// AttentionDecoder_60902636257372
// MI455X (gfx1250) — matrix operand order repaired and checked by reading, not compiled, not run on hardware
//
#include <hip/hip_runtime.h>
#include <hip/hip_bf16.h>
#include <stdint.h>

// ---------------- problem constants ----------------
#define BSZ    2048
#define TT     32
#define ENCD   512
#define DECD   256
#define ATTD   256
#define EMBD   64
#define NCLS   37
#define MAXL   10
#define XK     (EMBD + ENCD)     // 576
#define G3     (3 * DECD)        // 768
#define FCK    (DECD + ENCD)     // 768

// A-operand LDS pitches (halves): multiples of 8 -> 16B-aligned rows for ds_load_b128,
// pitch/2 mod 64 in {4,36} -> the 16 lanes' 4-dword reads tile all 64 LDS banks (conflict-free).
#define AP_H    264              // K=256
#define AP_ENC  520              // K=512
#define AP_X    584              // K=576
#define AP_FC   776              // K=768

typedef __attribute__((ext_vector_type(16))) _Float16 v16h;
typedef __attribute__((ext_vector_type(8)))  float    v8f;

union Frag { v16h v; uint4 q[2]; };

// ---- A fragment (16x32 f16) from LDS row-major [16, apitch] (apitch % 8 == 0) ----
// ISA layout: lanes 0-15 -> M=lane, K pairs {0..7,16..23}; lanes 16-31 -> same M, {8..15,24..31}
__device__ inline v16h frag_a_lds(const _Float16* A, int apitch, int kbase, int lane) {
  int m  = lane & 15;
  int hi = lane >> 4;                       // uint4 index 0 or 1
  Frag r;
  const uint4* row = (const uint4*)(A + m * apitch + kbase);   // 16B aligned
  r.q[0] = row[hi];                         // dwords 4*hi..4*hi+3  -> K = 8*hi + 0..7
  r.q[1] = row[2 + hi];                     // dwords 8+4*hi..      -> K = 16 + 8*hi + 0..7
  return r.v;
}

// ---- B fragment (32x16 f16) from global W^T row-major [N, K] ----
// ISA layout: lanes 0-15 hold K={0..7,16..23}, lanes 16-31 hold K={8..15,24..31}
__device__ inline v16h frag_b_g(const _Float16* WT, int kpitch, int n, int kbase, int lane) {
  Frag r;
  const uint4* row = (const uint4*)(WT + (size_t)n * kpitch + kbase); // 16B aligned
  int o = lane >> 4;                        // uint4 index 0 or 1
  r.q[0] = row[o];
  r.q[1] = row[o + 2];
  return r.v;
}

// ---- C[16,N](f32, LDS) = A[16,K](f16, LDS) * W[K,N] (as WT[N,K] f16 global) ----
// 8 waves split N into 16-wide tiles. All threads of the block must call (uniform EXEC).
__device__ inline void gemm16(const _Float16* A, int apitch, const _Float16* WT,
                              int K, int N, float* C, int cpitch, int tid) {
  int lane = tid & 31;
  int wave = tid >> 5;
  int hi8  = (lane >> 4) << 3;
  for (int nt = wave * 16; nt < N; nt += 16 * 8) {
    v8f acc = {};
    for (int kb = 0; kb < K; kb += 32) {
      v16h a = frag_a_lds(A, apitch, kb, lane);
      v16h b = frag_b_g(WT, K, nt + (lane & 15), kb, lane);
      acc = __builtin_amdgcn_wmma_f32_16x16x32_f16(false, a, false, b,
                                                   (short)0, acc, false, false);
    }
    int n = nt + (lane & 15);
#pragma unroll
    for (int r = 0; r < 8; ++r)
      C[(size_t)(r + hi8) * cpitch + n] = acc[r];
  }
}

// ---------------- weight convert + transpose: dst[n*K+k] = (f16) src[k*N+n] ----------------
__global__ void __launch_bounds__(256) k_transpose_f16(const float* __restrict__ src,
                                                       _Float16* __restrict__ dst,
                                                       int K, int N) {
  int i = blockIdx.x * 256 + threadIdx.x;
  if (i < K * N) {
    int n = i / K, k = i - n * K;
    dst[i] = (_Float16)src[(size_t)k * N + n];
  }
}

// ---------------- h0 = tanh(mean_T(enc) @ init_h_W + b) ----------------
__global__ void __launch_bounds__(256) k_h0(const float* __restrict__ enc,
                                            const _Float16* __restrict__ WT,   // [DEC, ENC]
                                            const float* __restrict__ bias,
                                            float* __restrict__ h) {
  __shared__ _Float16 A[16 * AP_ENC];
  __shared__ float    C[16 * DECD];
  int tid = threadIdx.x;
  size_t bbase = (size_t)blockIdx.x * 16;
  for (int b = 0; b < 16; ++b) {
    const float* p = enc + (bbase + b) * TT * ENCD;
    for (int e = tid; e < ENCD; e += 256) {
      float s = 0.f;
      for (int t = 0; t < TT; ++t) s += p[(size_t)t * ENCD + e];
      A[b * AP_ENC + e] = (_Float16)(s * (1.0f / 32.0f));
    }
  }
  __syncthreads();
  gemm16(A, AP_ENC, WT, ENCD, DECD, C, DECD, tid);
  __syncthreads();
  for (int i = tid; i < 16 * DECD; i += 256) {
    int j = i & (DECD - 1);
    h[bbase * DECD + i] = tanhf(C[i] + bias[j]);
  }
}

// ---------------- enc_proj = enc @ W_enc  ([B*T,512] x [512,256]) ----------------
__global__ void __launch_bounds__(256) k_encproj(const float* __restrict__ enc,
                                                 const _Float16* __restrict__ WT,  // [ATT, ENC]
                                                 float* __restrict__ ep) {
  __shared__ _Float16 A[16 * AP_ENC];
  __shared__ float    C[16 * ATTD];
  int tid = threadIdx.x;
  size_t rbase = (size_t)blockIdx.x * 16;
  for (int i = tid; i < 16 * ENCD; i += 256) {
    int r = i >> 9, e = i & (ENCD - 1);
    A[r * AP_ENC + e] = (_Float16)enc[(rbase + r) * ENCD + e];
  }
  __syncthreads();
  gemm16(A, AP_ENC, WT, ENCD, ATTD, C, ATTD, tid);
  __syncthreads();
  for (int i = tid; i < 16 * ATTD; i += 256) {
    int r = i >> 8, j = i & (ATTD - 1);
    ep[(rbase + r) * ATTD + j] = C[i];
  }
}

// ---------------- persistent per-tile decoder: 16 batch rows / WG, 10 steps ----------------
// LDS layout (bytes), all offsets 16B aligned
#define OFF_V     0                       // 256 f32   = 1024
#define OFF_BIH   1024                    // 768 f32   = 3072
#define OFF_BHH   4096                    // 768 f32   = 3072
#define OFF_FC1B  7168                    // 256 f32   = 1024
#define OFF_FC2B  8192                    // 37 f32 (pad 192)
#define OFF_EMBS  8384                    // 37*64 f32 = 9472
#define OFF_Y     17856                   // 16 i32    = 64
#define OFF_H     17920                   // 16*256 f32 = 16384
#define OFF_H16   34304                   // 16*264 f16 = 8448
#define OFF_DECP  42752                   // 16*256 f32 = 16384
#define OFF_EN    59136                   // 16*32  f32 = 2048
#define OFF_ATTN  61184                   // 16*32  f32 = 2048
#define OFF_X16   63232                   // 16*584 f16 = 18688
#define OFF_GI    81920                   // 16*768 f32 = 49152
#define OFF_GH    131072                  // 16*768 f32 = 49152
#define OFF_FCIN  180224                  // 16*776 f16 = 24832
#define OFF_FC1O  205056                  // 16*256 f32 = 16384
#define OFF_LOG   221440                  // 16*40  f32 = 2560
#define SMEM_BYTES 224000

__global__ void __launch_bounds__(256) k_decode(
    const float* __restrict__ enc, const float* __restrict__ embg,
    const _Float16* __restrict__ WdecT, const float* __restrict__ vg,
    const float* __restrict__ bih_g, const float* __restrict__ bhh_g,
    const _Float16* __restrict__ WihT, const _Float16* __restrict__ WhhT,
    const _Float16* __restrict__ fc1WT, const float* __restrict__ fc1b_g,
    const float* __restrict__ fc2W, const float* __restrict__ fc2b_g,
    const float* __restrict__ encproj, const float* __restrict__ hglob,
    float* __restrict__ out) {
  extern __shared__ char smem[];
  float*    v_s    = (float*)(smem + OFF_V);
  float*    bih_s  = (float*)(smem + OFF_BIH);
  float*    bhh_s  = (float*)(smem + OFF_BHH);
  float*    fc1b_s = (float*)(smem + OFF_FC1B);
  float*    fc2b_s = (float*)(smem + OFF_FC2B);
  float*    emb_s  = (float*)(smem + OFF_EMBS);
  int*      y_s    = (int*)  (smem + OFF_Y);
  float*    h_s    = (float*)(smem + OFF_H);
  _Float16* h16    = (_Float16*)(smem + OFF_H16);
  float*    decp   = (float*)(smem + OFF_DECP);
  float*    en_s   = (float*)(smem + OFF_EN);
  float*    at_s   = (float*)(smem + OFF_ATTN);
  _Float16* x16    = (_Float16*)(smem + OFF_X16);
  float*    gi_s   = (float*)(smem + OFF_GI);
  float*    gh_s   = (float*)(smem + OFF_GH);
  _Float16* fcin   = (_Float16*)(smem + OFF_FCIN);
  float*    fc1o   = (float*)(smem + OFF_FC1O);
  float*    log_s  = (float*)(smem + OFF_LOG);

  int tid = threadIdx.x;
  size_t bbase = (size_t)blockIdx.x * 16;

  // preload constants + state
  for (int i = tid; i < 256; i += 256) { v_s[i] = vg[i]; fc1b_s[i] = fc1b_g[i]; }
  for (int i = tid; i < G3;  i += 256) { bih_s[i] = bih_g[i]; bhh_s[i] = bhh_g[i]; }
  if (tid < NCLS) fc2b_s[tid] = fc2b_g[tid];
  for (int i = tid; i < NCLS * EMBD; i += 256) emb_s[i] = embg[i];
  for (int i = tid; i < 16 * DECD; i += 256) h_s[i] = hglob[bbase * DECD + i];
  if (tid < 16) y_s[tid] = 0;
  __syncthreads();

  for (int step = 0; step < MAXL; ++step) {
    // h -> f16 A operand
    for (int i = tid; i < 16 * DECD; i += 256) {
      int b = i >> 8, j = i & 255;
      h16[b * AP_H + j] = (_Float16)h_s[i];
    }
    __syncthreads();

    // dec_proj = h @ W_dec
    gemm16(h16, AP_H, WdecT, DECD, ATTD, decp, ATTD, tid);
    __syncthreads();

    // energy[b,t] = v . tanh(enc_proj[b,t,:] + dec_proj[b,:]) ; 8 lanes per (b,t)
    {
      int g = tid >> 3, j = tid & 7;
      for (int task = g; task < 16 * TT; task += 32) {
        int b = task >> 5, t = task & 31;
        const float* row = encproj + ((bbase + b) * TT + t) * ATTD;
        const float* dp  = decp + b * ATTD;
        float p = 0.f;
        for (int k = j; k < ATTD; k += 8) p += v_s[k] * tanhf(row[k] + dp[k]);
        p += __shfl_xor(p, 1, 8);
        p += __shfl_xor(p, 2, 8);
        p += __shfl_xor(p, 4, 8);
        if (j == 0) en_s[b * TT + t] = p;
      }
    }
    __syncthreads();

    // softmax over T per row
    if (tid < 16) {
      float mx = -1e30f;
      for (int t = 0; t < TT; ++t) mx = fmaxf(mx, en_s[tid * TT + t]);
      float s = 0.f;
      for (int t = 0; t < TT; ++t) { float e = __expf(en_s[tid * TT + t] - mx); at_s[tid * TT + t] = e; s += e; }
      float inv = 1.f / s;
      for (int t = 0; t < TT; ++t) at_s[tid * TT + t] *= inv;
    }
    __syncthreads();

    // context = attn . enc  -> f16 into x16[:,64:] and fcin[:,256:]
    for (int b = 0; b < 16; ++b) {
      const float* p = enc + (bbase + b) * TT * ENCD;
      for (int e = tid; e < ENCD; e += 256) {
        float acc = 0.f;
        for (int t = 0; t < TT; ++t) acc += at_s[b * TT + t] * p[(size_t)t * ENCD + e];
        _Float16 hv = (_Float16)acc;
        x16[b * AP_X + EMBD + e]  = hv;
        fcin[b * AP_FC + DECD + e] = hv;
      }
    }
    // embedding of previous y into x16[:,0:64]
    for (int i = tid; i < 16 * EMBD; i += 256) {
      int b = i >> 6, j = i & 63;
      x16[b * AP_X + j] = (_Float16)emb_s[y_s[b] * EMBD + j];
    }
    __syncthreads();

    // GRU gates: gi = x @ W_ih ; gh = h @ W_hh
    gemm16(x16, AP_X, WihT, XK, G3, gi_s, G3, tid);
    gemm16(h16, AP_H, WhhT, DECD, G3, gh_s, G3, tid);
    __syncthreads();

    // gate math (torch order r,z,n) -> h_new, f16 copy into fcin[:,0:256]
    for (int i = tid; i < 16 * DECD; i += 256) {
      int b = i >> 8, j = i & 255;
      float ir = gi_s[b * G3 + j]            + bih_s[j]            + gh_s[b * G3 + j]            + bhh_s[j];
      float iz = gi_s[b * G3 + DECD + j]     + bih_s[DECD + j]     + gh_s[b * G3 + DECD + j]     + bhh_s[DECD + j];
      float in = gi_s[b * G3 + 2 * DECD + j] + bih_s[2 * DECD + j];
      float hn = gh_s[b * G3 + 2 * DECD + j] + bhh_s[2 * DECD + j];
      float r  = 1.f / (1.f + __expf(-ir));
      float z  = 1.f / (1.f + __expf(-iz));
      float n  = tanhf(in + r * hn);
      float hnew = (1.f - z) * n + z * h_s[i];
      h_s[i] = hnew;
      fcin[b * AP_FC + j] = (_Float16)hnew;
    }
    __syncthreads();

    // fc1 + relu
    gemm16(fcin, AP_FC, fc1WT, FCK, DECD, fc1o, DECD, tid);
    __syncthreads();
    for (int i = tid; i < 16 * DECD; i += 256)
      fc1o[i] = fmaxf(fc1o[i] + fc1b_s[i & 255], 0.f);
    __syncthreads();

    // fc2 (K=256, N=37) in VALU + write outputs
    for (int c = tid; c < 16 * NCLS; c += 256) {
      int b = c / NCLS, j = c - b * NCLS;
      float acc = fc2b_s[j];
      const float* fr = fc1o + b * DECD;
      for (int k = 0; k < DECD; ++k) acc += fr[k] * fc2W[(size_t)k * NCLS + j];
      log_s[b * 40 + j] = acc;
      out[((bbase + b) * MAXL + step) * NCLS + j] = acc;
    }
    __syncthreads();

    // greedy argmax -> y
    if (tid < 16) {
      int best = 0; float bv = log_s[tid * 40];
      for (int j = 1; j < NCLS; ++j) {
        float vv = log_s[tid * 40 + j];
        if (vv > bv) { bv = vv; best = j; }
      }
      y_s[tid] = best;
    }
    __syncthreads();
  }
}

// ---------------- host launcher ----------------
extern "C" void kernel_launch(void* const* d_in, const int* in_sizes, int n_in,
                              void* d_out, int out_size, void* d_ws, size_t ws_size,
                              hipStream_t stream) {
  (void)in_sizes; (void)n_in; (void)out_size; (void)ws_size;
  const float* enc    = (const float*)d_in[0];
  const float* emb    = (const float*)d_in[1];
  const float* W_enc  = (const float*)d_in[2];
  const float* W_dec  = (const float*)d_in[3];
  const float* v      = (const float*)d_in[4];
  const float* initW  = (const float*)d_in[5];
  const float* initB  = (const float*)d_in[6];
  const float* W_ih   = (const float*)d_in[7];
  const float* b_ih   = (const float*)d_in[8];
  const float* W_hh   = (const float*)d_in[9];
  const float* b_hh   = (const float*)d_in[10];
  const float* fc1W   = (const float*)d_in[11];
  const float* fc1b   = (const float*)d_in[12];
  const float* fc2W   = (const float*)d_in[13];
  const float* fc2b   = (const float*)d_in[14];

  char* ws = (char*)d_ws;
  size_t off = 0;
  float* h_g  = (float*)(ws + off);  off += (size_t)BSZ * DECD * 4;
  float* ep_g = (float*)(ws + off);  off += (size_t)BSZ * TT * ATTD * 4;
  _Float16* WencT = (_Float16*)(ws + off); off += (size_t)ENCD * ATTD * 2;
  _Float16* WdecT = (_Float16*)(ws + off); off += (size_t)DECD * ATTD * 2;
  _Float16* initT = (_Float16*)(ws + off); off += (size_t)ENCD * DECD * 2;
  _Float16* WihT  = (_Float16*)(ws + off); off += (size_t)XK * G3 * 2;
  _Float16* WhhT  = (_Float16*)(ws + off); off += (size_t)DECD * G3 * 2;
  _Float16* fc1WT = (_Float16*)(ws + off); off += (size_t)FCK * DECD * 2;

  auto tgrid = [](int n) { return dim3((n + 255) / 256); };
  k_transpose_f16<<<tgrid(ENCD * ATTD), 256, 0, stream>>>(W_enc, WencT, ENCD, ATTD);
  k_transpose_f16<<<tgrid(DECD * ATTD), 256, 0, stream>>>(W_dec, WdecT, DECD, ATTD);
  k_transpose_f16<<<tgrid(ENCD * DECD), 256, 0, stream>>>(initW, initT, ENCD, DECD);
  k_transpose_f16<<<tgrid(XK * G3),     256, 0, stream>>>(W_ih,  WihT,  XK,   G3);
  k_transpose_f16<<<tgrid(DECD * G3),   256, 0, stream>>>(W_hh,  WhhT,  DECD, G3);
  k_transpose_f16<<<tgrid(FCK * DECD),  256, 0, stream>>>(fc1W,  fc1WT, FCK,  DECD);

  k_h0<<<BSZ / 16, 256, 0, stream>>>(enc, initT, initB, h_g);
  k_encproj<<<(BSZ * TT) / 16, 256, 0, stream>>>(enc, WencT, ep_g);

  k_decode<<<BSZ / 16, 256, SMEM_BYTES, stream>>>(
      enc, emb, WdecT, v, b_ih, b_hh, WihT, WhhT, fc1WT, fc1b, fc2W, fc2b,
      ep_g, h_g, (float*)d_out);
}
